// LayerConvol_85418309583433
// MI455X (gfx1250) — matrix operand order repaired and checked by reading, not compiled, not run on hardware
//
#include <hip/hip_runtime.h>
#include <hip/hip_bf16.h>
#include <math.h>

// ---------------------------------------------------------------------------
// Problem constants (match reference)
// ---------------------------------------------------------------------------
#define NN   8192
#define MM   8192
#define KF   512      // IN_F == OUT_F == 512
#define NEG_BIG_F (-4294967295.0f)          // -2^32 + 1
#define SOFTMAX_SCALE (1.4426950408889634f / 0.07f)   // log2(e) / TEMPERATURE

#define QSTR 520      // padded LDS row stride (halves) for Q: banks 4*lane
#define PSTR 136      // padded LDS row stride (halves) for P tile

// sched_barrier masks (bit = instruction class that MAY cross the barrier):
//  0x2 VALU, 0x4 SALU, 0x20 VMEM read, 0x100 DS read; 0x8 (WMMA) never set.
#define SB_PRE  0x006   // between loads and WMMAs: loads may NOT sink past
#define SB_POST 0x126   // after WMMAs: later loads MAY hoist over the WMMAs

typedef __attribute__((ext_vector_type(16))) _Float16 v16h;
typedef __attribute__((ext_vector_type(8)))  float    v8f;

// Load a 16-half WMMA fragment from two 16-byte chunks.
__device__ __forceinline__ v16h ldfrag(const _Float16* p0, const _Float16* p1) {
    v16h r;
    __builtin_memcpy(&r, p0, 16);
    __builtin_memcpy(reinterpret_cast<char*>(&r) + 16, p1, 16);
    return r;
}

__device__ __forceinline__ v8f wmma_f16(v16h a, v16h b, v8f c) {
    return __builtin_amdgcn_wmma_f32_16x16x32_f16(false, a, false, b,
                                                  (short)0, c, false, false);
}

// ---------------------------------------------------------------------------
// gfx1250 async global -> LDS copy (ASYNCcnt-tracked DMA), inline asm so it
// works on both ROCm-7.2 and amdgpu-toolchain header sets.
// ---------------------------------------------------------------------------
__device__ __forceinline__ void async_copy_b128(unsigned lds_off, const void* gptr) {
    asm volatile("global_load_async_to_lds_b128 %0, %1, off"
                 :: "v"(lds_off), "v"(gptr)
                 : "memory");
}
__device__ __forceinline__ void wait_async_le4() {
    asm volatile("s_wait_asynccnt 0x4" ::: "memory");
}
__device__ __forceinline__ void wait_async_0() {
    asm volatile("s_wait_asynccnt 0x0" ::: "memory");
}

// Issue this wave's adj slice (16 rows x 32 ints = 2 KB) as 4 async B128 ops.
__device__ __forceinline__ void issue_adj_async(const int* __restrict__ adj,
                                                int rowbase, int jcolbase,
                                                int* lds_dst,   // [16][32] ints
                                                int lane) {
    #pragma unroll
    for (int q = 0; q < 4; ++q) {
        const int c    = q * 32 + lane;   // 0..127 chunk id
        const int row  = c >> 3;          // 0..15
        const int col4 = (c & 7) * 4;     // 0,4,...,28
        unsigned loff = (unsigned)(uintptr_t)(lds_dst + row * 32 + col4);
        const void* g = adj + (size_t)(rowbase + row) * MM + jcolbase + col4;
        async_copy_b128(loff, g);
    }
}

// ---------------------------------------------------------------------------
// Kernel 1: fp32 -> f16 convert, optionally also writing a transposed copy.
// blockDim = (32, 8); each block handles a 32x32 tile.
// ---------------------------------------------------------------------------
__global__ void cvt_kernel(const float* __restrict__ src,
                           _Float16* __restrict__ dstP,   // [R][C] or null
                           _Float16* __restrict__ dstT,   // [C][R] or null
                           int R, int C) {
    __shared__ _Float16 tile[32][33];
    const int tc = blockIdx.x * 32;
    const int tr = blockIdx.y * 32;
    const int tx = threadIdx.x;          // 0..31
    const int ty = threadIdx.y;          // 0..7
    for (int yy = ty; yy < 32; yy += 8) {
        float v = src[(size_t)(tr + yy) * C + tc + tx];
        _Float16 h = (_Float16)v;
        tile[yy][tx] = h;
        if (dstP) dstP[(size_t)(tr + yy) * C + tc + tx] = h;
    }
    __syncthreads();
    if (dstT) {
        for (int yy = ty; yy < 32; yy += 8) {
            dstT[(size_t)(tc + yy) * R + tr + tx] = tile[tx][yy];
        }
    }
}

// ---------------------------------------------------------------------------
// Kernel 2: proj = anchor @ W   (f16 in, f32 accumulate, f16 out)
// 128 threads = 4 waves; block does 16 rows x 512 cols; wave w does 128 cols.
// Per k-step: [1 A-frag + 8 B-frag loads] | barrier(0x6) | [8 WMMAs] | 0x126.
// ---------------------------------------------------------------------------
__global__ __launch_bounds__(128)
void gemm1_kernel(const _Float16* __restrict__ A,   // anchor_f16 [NN][KF]
                  const _Float16* __restrict__ BT,  // W^T f16 [out][in]
                  _Float16* __restrict__ P) {       // proj_f16 [NN][KF]
    const int lane = threadIdx.x & 31;
    const int wave = threadIdx.x >> 5;
    const int lo = lane & 15;
    const int hi = lane >> 4;
    const int rb = blockIdx.x * 16;
    const int cb = wave * 128;

    v8f acc[8] = {};
    const _Float16* arow = A + (size_t)(rb + lo) * KF + (hi ? 16 : 0);
    const _Float16* brow = BT + (size_t)(cb + lo) * KF + (hi ? 16 : 0);
    for (int k0 = 0; k0 < KF; k0 += 32) {
        v16h a = ldfrag(arow + k0, arow + k0 + 8);
        v16h b[8];
        #pragma unroll
        for (int n = 0; n < 8; ++n) {
            const _Float16* pb = brow + (size_t)(n * 16) * KF + k0;
            b[n] = ldfrag(pb, pb + 8);
        }
        __builtin_amdgcn_sched_barrier(SB_PRE);
        #pragma unroll
        for (int n = 0; n < 8; ++n)
            acc[n] = wmma_f16(a, b[n], acc[n]);
        __builtin_amdgcn_sched_barrier(SB_POST);
    }
    #pragma unroll
    for (int n = 0; n < 8; ++n)
        #pragma unroll
        for (int i = 0; i < 8; ++i)
            P[(size_t)(rb + i + 8 * hi) * KF + cb + n * 16 + lo] =
                (_Float16)acc[n][i];
}

// ---------------------------------------------------------------------------
// Kernel 3: fused masked-softmax attention (flash-attention style).
// Block = 128 threads (4 waves) owns 16 output rows.
// adj tiles are DMA'd global->LDS one tile ahead (double buffered, per-wave
// slices, ASYNCcnt-tracked) so the 256 MB HBM stream hides under the WMMAs.
// ---------------------------------------------------------------------------
__global__ __launch_bounds__(128)
void attn_kernel(const _Float16* __restrict__ Q,    // proj_f16 [NN][KF]
                 const _Float16* __restrict__ Krm,  // input_f16 [MM][KF]
                 const _Float16* __restrict__ Vt,   // inputT_f16 [KF][MM]
                 const int* __restrict__ adj,       // [NN][MM]
                 float* __restrict__ out) {         // [NN][KF]
    __shared__ __align__(16) _Float16 lds_q[16 * QSTR];   // padded: no bank cfl
    __shared__ __align__(16) _Float16 lds_p[16 * PSTR];   // padded: no bank cfl
    __shared__ __align__(16) int lds_adj[2][4][16][32];   // 16 KB (dbl buf)
    __shared__ float lds_rmax[4][16];
    __shared__ float lds_rsum[4][16];

    const int lane = threadIdx.x & 31;
    const int wave = threadIdx.x >> 5;
    const int lo = lane & 15;
    const int hi = lane >> 4;
    const int rowbase = blockIdx.x * 16;
    const int cbase = wave * 128;        // output-column slice of this wave

    // Stage Q (16 rows x 512 halves) into padded LDS rows with 16B vectors.
    {
        const float4* src = (const float4*)(Q + (size_t)rowbase * KF);
        for (int t = threadIdx.x; t < 1024; t += 128) {
            const int r = t >> 6;        // row 0..15
            const int c = t & 63;        // float4 chunk within row
            *((float4*)(lds_q + r * QSTR) + c) = src[t];
        }
    }
    __syncthreads();

    float mrow[8], lrow[8];
    v8f oacc[8] = {};
    #pragma unroll
    for (int i = 0; i < 8; ++i) { mrow[i] = -3.0e38f; lrow[i] = 0.0f; }

    // Prime the adj pipeline: tile 0 into buffer 0.
    issue_adj_async(adj, rowbase, 0 + wave * 32, &lds_adj[0][wave][0][0], lane);
    int bufc = 0;

    for (int jt = 0; jt < MM; jt += 128) {
        const int j0 = jt + wave * 32;
        const bool have_next = (jt + 128) < MM;

        // Kick off the DMA for the NEXT adj tile before doing any math.
        if (have_next)
            issue_adj_async(adj, rowbase, jt + 128 + wave * 32,
                            &lds_adj[bufc ^ 1][wave][0][0], lane);

        // ---- S = Q @ K^T for this wave's 32 columns --------------------
        v8f s0 = {}, s1 = {};
        {
            const _Float16* qb  = lds_q + lo * QSTR + (hi ? 16 : 0);
            const _Float16* kp0 = Krm + (size_t)(j0 + lo) * KF + (hi ? 16 : 0);
            const _Float16* kp1 = kp0 + (size_t)16 * KF;
            for (int k = 0; k < KF; k += 64) {
                v16h a0  = ldfrag(qb + k,       qb + k + 8);
                v16h a1  = ldfrag(qb + k + 32,  qb + k + 40);
                v16h b00 = ldfrag(kp0 + k,      kp0 + k + 8);
                v16h b10 = ldfrag(kp1 + k,      kp1 + k + 8);
                v16h b01 = ldfrag(kp0 + k + 32, kp0 + k + 40);
                v16h b11 = ldfrag(kp1 + k + 32, kp1 + k + 40);
                __builtin_amdgcn_sched_barrier(SB_PRE);
                s0 = wmma_f16(a0, b00, s0);
                s1 = wmma_f16(a0, b10, s1);
                s0 = wmma_f16(a1, b01, s0);
                s1 = wmma_f16(a1, b11, s1);
                __builtin_amdgcn_sched_barrier(SB_POST);
            }
        }

        // ---- adjacency mask from the LDS tile DMA'd last iteration -----
        if (have_next) wait_async_le4();   // oldest 4 (this tile) complete
        else           wait_async_0();
        float v0[8], v1[8];
        #pragma unroll
        for (int i = 0; i < 8; ++i) {
            const int rr = i + 8 * hi;
            v0[i] = (lds_adj[bufc][wave][rr][lo]      > 0) ? s0[i] : NEG_BIG_F;
            v1[i] = (lds_adj[bufc][wave][rr][16 + lo] > 0) ? s1[i] : NEG_BIG_F;
        }

        // ---- per-wave row max over 32 cols ------------------------------
        float t[8];
        #pragma unroll
        for (int i = 0; i < 8; ++i) t[i] = fmaxf(v0[i], v1[i]);
        #pragma unroll
        for (int m = 1; m < 16; m <<= 1)
            #pragma unroll
            for (int i = 0; i < 8; ++i)
                t[i] = fmaxf(t[i], __shfl_xor(t[i], m, 32));
        if (lo == 0)
            #pragma unroll
            for (int i = 0; i < 8; ++i) lds_rmax[wave][i + 8 * hi] = t[i];
        __syncthreads();

        // ---- global tile max, online-softmax rescale factor -------------
        float newm[8], pscale[8];
        #pragma unroll
        for (int i = 0; i < 8; ++i) {
            const int rr = i + 8 * hi;
            float gm = fmaxf(fmaxf(lds_rmax[0][rr], lds_rmax[1][rr]),
                             fmaxf(lds_rmax[2][rr], lds_rmax[3][rr]));
            float nm = fmaxf(mrow[i], gm);
            pscale[i] = exp2f((mrow[i] - nm) * SOFTMAX_SCALE);
            newm[i] = nm;
            mrow[i] = nm;
        }

        // ---- P = exp2((s - m) * scale), stash f16 P tile in LDS ---------
        float ps[8];
        #pragma unroll
        for (int i = 0; i < 8; ++i) {
            float p0 = exp2f((v0[i] - newm[i]) * SOFTMAX_SCALE);
            float p1 = exp2f((v1[i] - newm[i]) * SOFTMAX_SCALE);
            ps[i] = p0 + p1;
            const int pr = (i + 8 * hi) * PSTR + wave * 32 + lo;
            lds_p[pr]      = (_Float16)p0;
            lds_p[pr + 16] = (_Float16)p1;
        }
        #pragma unroll
        for (int m = 1; m < 16; m <<= 1)
            #pragma unroll
            for (int i = 0; i < 8; ++i)
                ps[i] += __shfl_xor(ps[i], m, 32);
        if (lo == 0)
            #pragma unroll
            for (int i = 0; i < 8; ++i) lds_rsum[wave][i + 8 * hi] = ps[i];
        __syncthreads();

        #pragma unroll
        for (int i = 0; i < 8; ++i) {
            const int rr = i + 8 * hi;
            float s = (lds_rsum[0][rr] + lds_rsum[1][rr]) +
                      (lds_rsum[2][rr] + lds_rsum[3][rr]);
            lrow[i] = lrow[i] * pscale[i] + s;
        }
        #pragma unroll
        for (int n = 0; n < 8; ++n)
            #pragma unroll
            for (int i = 0; i < 8; ++i)
                oacc[n][i] *= pscale[i];

        // ---- O += P @ V  (loads grouped ahead of the 8 WMMAs) -----------
        #pragma unroll
        for (int kp = 0; kp < 128; kp += 32) {
            const _Float16* pa = lds_p + lo * PSTR + kp + (hi ? 16 : 0);
            v16h a = ldfrag(pa, pa + 8);
            v16h b[8];
            #pragma unroll
            for (int n = 0; n < 8; ++n) {
                const _Float16* pv = Vt +
                    (size_t)(cbase + n * 16 + lo) * MM + jt + kp + (hi ? 16 : 0);
                b[n] = ldfrag(pv, pv + 8);
            }
            __builtin_amdgcn_sched_barrier(SB_PRE);
            #pragma unroll
            for (int n = 0; n < 8; ++n)
                oacc[n] = wmma_f16(a, b[n], oacc[n]);
            __builtin_amdgcn_sched_barrier(SB_POST);
        }
        __syncthreads();   // lds_p / lds_rmax / lds_rsum reused next iter
        bufc ^= 1;
    }

    // ---- finalize: divide by denominator, non-temporal fp32 stores ------
    #pragma unroll
    for (int n = 0; n < 8; ++n)
        #pragma unroll
        for (int i = 0; i < 8; ++i)
            __builtin_nontemporal_store(
                oacc[n][i] / lrow[i],
                out + (size_t)(rowbase + i + 8 * hi) * KF + cbase + n * 16 + lo);
}

// ---------------------------------------------------------------------------
// Host launcher
// ---------------------------------------------------------------------------
extern "C" void kernel_launch(void* const* d_in, const int* in_sizes, int n_in,
                              void* d_out, int out_size, void* d_ws, size_t ws_size,
                              hipStream_t stream) {
    (void)in_sizes; (void)n_in; (void)out_size; (void)ws_size;
    const float* xx  = (const float*)d_in[0];   // [NN][KF]
    const float* inp = (const float*)d_in[1];   // [MM][KF]
    const int*   adj = (const int*)d_in[2];     // [NN][MM]
    const float* w   = (const float*)d_in[3];   // [KF][KF]
    float* out = (float*)d_out;                 // [NN][KF]

    char* ws = (char*)d_ws;
    _Float16* inp_h  = (_Float16*)(ws);                        // 8 MB
    _Float16* inpT_h = (_Float16*)(ws + ((size_t)8  << 20));   // 8 MB
    _Float16* anc_h  = (_Float16*)(ws + ((size_t)16 << 20));   // 8 MB
    _Float16* wT_h   = (_Float16*)(ws + ((size_t)24 << 20));   // 0.5 MB
    _Float16* proj_h = (_Float16*)(ws + ((size_t)25 << 20));   // 8 MB

    dim3 tb(32, 8);
    cvt_kernel<<<dim3(KF / 32, NN / 32), tb, 0, stream>>>(inp, inp_h, inpT_h, NN, KF);
    cvt_kernel<<<dim3(KF / 32, NN / 32), tb, 0, stream>>>(xx, anc_h, nullptr, NN, KF);
    cvt_kernel<<<dim3(KF / 32, KF / 32), tb, 0, stream>>>(w, nullptr, wT_h, KF, KF);

    gemm1_kernel<<<NN / 16, 128, 0, stream>>>(anc_h, wT_h, proj_h);
    attn_kernel<<<NN / 16, 128, 0, stream>>>(proj_h, inp_h, inpT_h, adj, out);
}
